// ChildSumTreeLSTM_24352464569160
// MI455X (gfx1250) — matrix operand order repaired and checked by reading, not compiled, not run on hardware
//
#include <hip/hip_runtime.h>
#include <hip/hip_bf16.h>

// ChildSumTreeLSTM on a complete binary tree, N = 131071 = 2^17 - 1 nodes.
// Pipeline:
//   1. convert_weights : f32 -> bf16 copies of W matrices in ws
//   2. gemm_input      : Y[N][1024] = X @ [W_ioux;W_fx]^T + bias   (bf16 WMMA, f32 accum)
//                        M=32 rows/block so each B fragment is reused by 2 WMMAs
//                        (halves L2 weight traffic vs M=16).
//   3. leaf_level      : gate math for the 65536 leaves
//   4. tree_level x16  : per level, 32 parents/block: child-sum GEMM (256->768) +
//                        per-child f GEMM (256->256) via WMMA, then gate math;
//                        root writes d_out.
//
// ws layout (bytes):  Y: N*1024*4 | H: N*256*4 | C: N*256*4 | bf16 weights (~1 MB)

#define N_NODES 131071
#define IN_DIM  256
#define MEM_DIM 256
#define YPITCH  1024

typedef __attribute__((ext_vector_type(16))) __bf16 v16bf;
typedef __attribute__((ext_vector_type(8)))  __bf16 v8bf;
typedef __attribute__((ext_vector_type(8)))  float  v8f;

__device__ __forceinline__ __bf16 f2bf(float f) {
  unsigned u = __builtin_bit_cast(unsigned, f);
  unsigned r = (u + 0x7FFFu + ((u >> 16) & 1u)) >> 16;   // round-to-nearest-even
  unsigned short s = (unsigned short)r;
  return __builtin_bit_cast(__bf16, s);
}

__device__ __forceinline__ v8f wmma_bf16(v16bf a, v16bf b, v8f c) {
  return __builtin_amdgcn_wmma_f32_16x16x32_bf16(false, a, false, b, (short)0, c,
                                                 false, false);
}

// A-fragment (16x32 bf16) from an LDS tile, contiguous k order (same as load_b_w):
// lane L holds row M=L%16; lanes<16: K in kBase+0..15,
// lanes>=16: K in kBase+16..31.
__device__ __forceinline__ v16bf load_a_lds(const __bf16* base, int pitch, int lane,
                                            int kBase) {
  int row = lane & 15;
  int off = kBase + ((lane < 16) ? 0 : 16);
  const __bf16* p = base + row * pitch + off;
  v8bf lo = *(const v8bf*)p;
  v8bf hi = *(const v8bf*)(p + 8);
  return __builtin_shufflevector(lo, hi, 0, 1, 2, 3, 4, 5, 6, 7, 8, 9, 10, 11, 12,
                                 13, 14, 15);
}

// B-fragment (32x16 bf16) = W^T tile: lane L holds column N=L%16 (a W row),
// K = kBase + (L<16 ? 0..15 : 16..31) -> one contiguous 32-byte load per lane.
__device__ __forceinline__ v16bf load_b_w(const __bf16* W, int outBase, int lane,
                                          int kBase) {
  int col = outBase + (lane & 15);
  int off = kBase + ((lane < 16) ? 0 : 16);
  return *(const v16bf*)(W + (size_t)col * IN_DIM + off);
}

__device__ __forceinline__ float sigmoidf_(float x) {
  return 1.0f / (1.0f + __expf(-x));
}

// ---------------------------------------------------------------- weights->bf16
__global__ void convert_weights(const float* __restrict__ W_ioux,
                                const float* __restrict__ W_fx,
                                const float* __restrict__ W_iouh,
                                const float* __restrict__ W_fh,
                                __bf16* __restrict__ WXB,
                                __bf16* __restrict__ WIOUHB,
                                __bf16* __restrict__ WFHB) {
  int idx = blockIdx.x * blockDim.x + threadIdx.x;
  if (idx < 1024 * 256) {                       // concat [W_ioux ; W_fx] rows
    int o = idx >> 8, i = idx & 255;
    float v = (o < 768) ? W_ioux[o * 256 + i] : W_fx[(o - 768) * 256 + i];
    WXB[idx] = f2bf(v);
  } else if (idx < 1024 * 256 + 768 * 256) {
    int j = idx - 1024 * 256;
    WIOUHB[j] = f2bf(W_iouh[j]);
  } else if (idx < 1024 * 256 + 768 * 256 + 256 * 256) {
    int j = idx - (1024 * 256 + 768 * 256);
    WFHB[j] = f2bf(W_fh[j]);
  }
}

// ------------------------------------------------- Y = X @ [W_ioux;W_fx]^T + b
// Block: 32 rows x 1024 cols; 8 waves, each wave owns 8 N-tiles x 2 M-tiles.
// B fragment loaded once per (k, N-tile), fed to 2 WMMAs (A row-halves).
__global__ void __launch_bounds__(256) gemm_input(const float* __restrict__ X,
                                                  const __bf16* __restrict__ WXB,
                                                  const float* __restrict__ b_ioux,
                                                  const float* __restrict__ b_fx,
                                                  float* __restrict__ Y) {
  __shared__ __align__(16) __bf16 sA[32 * 40];  // 32 rows x 32 k, pitch 40
  const int tid = threadIdx.x;
  const int wave = tid >> 5, lane = tid & 31;
  const int rb = blockIdx.x * 32;

  v8f acc[16] = {};                             // [half*8 + t]
  for (int ks = 0; ks < 8; ++ks) {
    const int kBase = ks * 32;
#pragma unroll
    for (int t = 0; t < 4; ++t) {               // stage 32x32 f32 -> bf16 LDS
      int idx = t * 256 + tid;
      int r = idx >> 5, kk = idx & 31;
      int n = rb + r;
      float v = (n < N_NODES) ? X[(size_t)n * IN_DIM + kBase + kk] : 0.0f;
      sA[r * 40 + kk] = f2bf(v);
    }
    __syncthreads();
    v16bf a0 = load_a_lds(sA, 40, lane, 0);             // rows 0..15
    v16bf a1 = load_a_lds(sA + 16 * 40, 40, lane, 0);   // rows 16..31
#pragma unroll
    for (int t = 0; t < 8; ++t) {
      v16bf b = load_b_w(WXB, wave * 128 + t * 16, lane, kBase);
      acc[t]     = wmma_bf16(a0, b, acc[t]);
      acc[8 + t] = wmma_bf16(a1, b, acc[8 + t]);
    }
    __syncthreads();
  }
#pragma unroll
  for (int t = 0; t < 8; ++t) {                 // bias + store (C layout)
    int col = wave * 128 + t * 16 + (lane & 15);
    float bias = (col < 768) ? b_ioux[col] : b_fx[col - 768];
#pragma unroll
    for (int half = 0; half < 2; ++half) {
#pragma unroll
      for (int r = 0; r < 8; ++r) {
        int m = half * 16 + r + ((lane < 16) ? 0 : 8);
        int n = rb + m;
        if (n < N_NODES) Y[(size_t)n * YPITCH + col] = acc[half * 8 + t][r] + bias;
      }
    }
  }
}

// ------------------------------------------------------------------ leaf level
__global__ void leaf_level(const float* __restrict__ Y,
                           const float* __restrict__ b_iouh,
                           float* __restrict__ Hbuf, float* __restrict__ Cbuf) {
  const int start = 65535;                      // 2^16 - 1
  int idx = blockIdx.x * blockDim.x + threadIdx.x;   // exactly 65536*256 threads
  int j = idx >> 8, m = idx & 255;
  int n = start + j;
  const float* y = Y + (size_t)n * YPITCH;
  float gi = y[m] + b_iouh[m];
  float go = y[256 + m] + b_iouh[256 + m];
  float gu = y[512 + m] + b_iouh[512 + m];
  float c = sigmoidf_(gi) * tanhf(gu);
  float h = sigmoidf_(go) * tanhf(c);
  Cbuf[(size_t)n * MEM_DIM + m] = c;
  Hbuf[(size_t)n * MEM_DIM + m] = h;
}

// ---------------------------------------------------------------- inner levels
// Block = 32 parent nodes. Children of nodes [nb, nb+32) are the contiguous
// rows [2*nb+1, 2*nb+65). GEMM1: childsum(32x256) @ W_iouh^T -> 32x768.
// GEMM2: children(64x256) @ W_fh^T -> 64x256. Then fused gate math.
__global__ void __launch_bounds__(256) tree_level(
    const float* __restrict__ Y, const __bf16* __restrict__ WIOUHB,
    const __bf16* __restrict__ WFHB, const float* __restrict__ b_iouh,
    const float* __restrict__ b_fh, float* __restrict__ Hbuf,
    float* __restrict__ Cbuf, float* __restrict__ out, int start, int count) {
  extern __shared__ char smem[];
  __bf16* sSum = (__bf16*)smem;                                   // 32 x 264
  __bf16* sCh  = (__bf16*)(smem + 32 * 264 * 2);                  // 64 x 264
  float*  sIou = (float*)(smem + 32 * 264 * 2 + 64 * 264 * 2);    // 32 x 768
  float*  sFh  = (float*)(smem + 32 * 264 * 2 + 64 * 264 * 2 + 32 * 768 * 4); // 64x256

  const int tid = threadIdx.x, wave = tid >> 5, lane = tid & 31;
  const int nb = start + blockIdx.x * 32;
  const int limit = start + count;

  // Stage child h rows (bf16) and child-sum rows (f32 add, then bf16).
#pragma unroll 4
  for (int t = 0; t < 64; ++t) {
    int idx = t * 256 + tid;
    int r = idx >> 8, m = idx & 255;
    int parent = nb + (r >> 1);
    float v = (parent < limit) ? Hbuf[(size_t)(2 * nb + 1 + r) * MEM_DIM + m] : 0.0f;
    sCh[r * 264 + m] = f2bf(v);
  }
#pragma unroll 4
  for (int t = 0; t < 32; ++t) {
    int idx = t * 256 + tid;
    int j = idx >> 8, m = idx & 255;
    int n = nb + j;
    float v = 0.0f;
    if (n < limit)
      v = Hbuf[(size_t)(2 * n + 1) * MEM_DIM + m] +
          Hbuf[(size_t)(2 * n + 2) * MEM_DIM + m];
    sSum[j * 264 + m] = f2bf(v);
  }
  __syncthreads();

  {  // GEMM1: 48 N-tiles x 2 M-tiles = 96 tiles -> 6 N-tiles x 2 M per wave
    v8f acc[12] = {};                           // [half*6 + t]
    for (int ks = 0; ks < 8; ++ks) {
      int kBase = ks * 32;
      v16bf a0 = load_a_lds(sSum, 264, lane, kBase);            // rows 0..15
      v16bf a1 = load_a_lds(sSum + 16 * 264, 264, lane, kBase); // rows 16..31
#pragma unroll
      for (int t = 0; t < 6; ++t) {
        v16bf b = load_b_w(WIOUHB, (wave * 6 + t) * 16, lane, kBase);
        acc[t]     = wmma_bf16(a0, b, acc[t]);
        acc[6 + t] = wmma_bf16(a1, b, acc[6 + t]);
      }
    }
#pragma unroll
    for (int t = 0; t < 6; ++t) {
      int col = (wave * 6 + t) * 16 + (lane & 15);
#pragma unroll
      for (int half = 0; half < 2; ++half) {
#pragma unroll
        for (int r = 0; r < 8; ++r) {
          int row = half * 16 + r + ((lane < 16) ? 0 : 8);
          sIou[row * 768 + col] = acc[half * 6 + t][r];
        }
      }
    }
  }
  {  // GEMM2: 4 M-tiles x 16 N-tiles = 64 tiles; wave -> 2 M-tiles x 4 N-tiles
    v8f acc[8] = {};                            // [half*4 + t]
    const int mt0 = (wave >> 2) * 2;            // waves 0-3 -> rows 0..31, 4-7 -> 32..63
    for (int ks = 0; ks < 8; ++ks) {
      int kBase = ks * 32;
      v16bf a0 = load_a_lds(sCh + mt0 * 16 * 264, 264, lane, kBase);
      v16bf a1 = load_a_lds(sCh + (mt0 + 1) * 16 * 264, 264, lane, kBase);
#pragma unroll
      for (int t = 0; t < 4; ++t) {
        v16bf b = load_b_w(WFHB, ((wave & 3) * 4 + t) * 16, lane, kBase);
        acc[t]     = wmma_bf16(a0, b, acc[t]);
        acc[4 + t] = wmma_bf16(a1, b, acc[4 + t]);
      }
    }
#pragma unroll
    for (int t = 0; t < 4; ++t) {
      int col = ((wave & 3) * 4 + t) * 16 + (lane & 15);
#pragma unroll
      for (int half = 0; half < 2; ++half) {
#pragma unroll
        for (int r = 0; r < 8; ++r) {
          int row = (mt0 + half) * 16 + r + ((lane < 16) ? 0 : 8);
          sFh[row * 256 + col] = acc[half * 4 + t][r];
        }
      }
    }
  }
  __syncthreads();

  // Fused gate math: 32 nodes x 256 dims.
#pragma unroll 4
  for (int t = 0; t < 32; ++t) {
    int idx = t * 256 + tid;
    int j = idx >> 8, m = idx & 255;
    int n = nb + j;
    if (n >= limit) continue;
    const float* y = Y + (size_t)n * YPITCH;
    float gi = y[m] + sIou[j * 768 + m] + b_iouh[m];
    float go = y[256 + m] + sIou[j * 768 + 256 + m] + b_iouh[256 + m];
    float gu = y[512 + m] + sIou[j * 768 + 512 + m] + b_iouh[512 + m];
    float fxv = y[768 + m];
    float fl = sigmoidf_(sFh[(2 * j) * 256 + m] + b_fh[m] + fxv);
    float fr = sigmoidf_(sFh[(2 * j + 1) * 256 + m] + b_fh[m] + fxv);
    float cl = Cbuf[(size_t)(2 * n + 1) * MEM_DIM + m];
    float cr = Cbuf[(size_t)(2 * n + 2) * MEM_DIM + m];
    float cn = sigmoidf_(gi) * tanhf(gu) + fl * cl + fr * cr;
    float hn = sigmoidf_(go) * tanhf(cn);
    Cbuf[(size_t)n * MEM_DIM + m] = cn;
    Hbuf[(size_t)n * MEM_DIM + m] = hn;
    if (out && n == 0) { out[m] = cn; out[256 + m] = hn; }  // root -> d_out
  }
}

// ---------------------------------------------------------------------- launch
extern "C" void kernel_launch(void* const* d_in, const int* in_sizes, int n_in,
                              void* d_out, int out_size, void* d_ws, size_t ws_size,
                              hipStream_t stream) {
  const float* X      = (const float*)d_in[0];
  const float* W_ioux = (const float*)d_in[1];
  const float* b_ioux = (const float*)d_in[2];
  const float* W_iouh = (const float*)d_in[3];
  const float* b_iouh = (const float*)d_in[4];
  const float* W_fx   = (const float*)d_in[5];
  const float* b_fx   = (const float*)d_in[6];
  const float* W_fh   = (const float*)d_in[7];
  const float* b_fh   = (const float*)d_in[8];

  char* ws = (char*)d_ws;
  size_t offY     = 0;
  size_t offH     = offY + (size_t)N_NODES * YPITCH * 4;
  size_t offC     = offH + (size_t)N_NODES * MEM_DIM * 4;
  size_t offWXB   = offC + (size_t)N_NODES * MEM_DIM * 4;
  size_t offWIOUH = offWXB + (size_t)1024 * 256 * 2;
  size_t offWFH   = offWIOUH + (size_t)768 * 256 * 2;

  float*  Y      = (float*)(ws + offY);
  float*  Hbuf   = (float*)(ws + offH);
  float*  Cbuf   = (float*)(ws + offC);
  __bf16* WXB    = (__bf16*)(ws + offWXB);
  __bf16* WIOUHB = (__bf16*)(ws + offWIOUH);
  __bf16* WFHB   = (__bf16*)(ws + offWFH);

  hipLaunchKernelGGL(convert_weights, dim3(2048), dim3(256), 0, stream,
                     W_ioux, W_fx, W_iouh, W_fh, WXB, WIOUHB, WFHB);
  hipLaunchKernelGGL(gemm_input, dim3((N_NODES + 31) / 32), dim3(256), 0, stream,
                     X, WXB, b_ioux, b_fx, Y);
  hipLaunchKernelGGL(leaf_level, dim3(65536), dim3(256), 0, stream,
                     Y, b_iouh, Hbuf, Cbuf);

  const size_t smemBytes = (size_t)32 * 264 * 2 + (size_t)64 * 264 * 2 +
                           (size_t)32 * 768 * 4 + (size_t)64 * 256 * 4;  // 214528 B
  for (int lvl = 15; lvl >= 0; --lvl) {
    int start = (1 << lvl) - 1;
    int count = (1 << lvl);
    int grid = (count + 31) / 32;
    float* out = (lvl == 0) ? (float*)d_out : nullptr;
    hipLaunchKernelGGL(tree_level, dim3(grid), dim3(256), smemBytes, stream,
                       Y, WIOUHB, WFHB, b_iouh, b_fh, Hbuf, Cbuf, out, start, count);
  }
}
